// MultiRelationGATBlock_1211180778452
// MI455X (gfx1250) — compile-verified, matrix operand order checked by reading, not run on hardware
//
#include <hip/hip_runtime.h>
#include <hip/hip_bf16.h>

#define EPS        1e-5f
#define NEG_SLOPE  0.2f
#define USE_ASYNC  1

typedef __bf16 bf16;
typedef __attribute__((ext_vector_type(16))) bf16  v16bf;
typedef __attribute__((ext_vector_type(8)))  bf16  v8bf;
typedef __attribute__((ext_vector_type(8)))  float v8f;

// ---------------------------------------------------------------------------
// fp32 -> bf16 conversion (grid-stride) -- only needed for h
// ---------------------------------------------------------------------------
__global__ void cvt_f32_bf16_kernel(const float* __restrict__ src,
                                    bf16* __restrict__ dst, long n) {
    long i = (long)blockIdx.x * blockDim.x + threadIdx.x;
    long stride = (long)gridDim.x * blockDim.x;
    for (; i < n; i += stride) dst[i] = (bf16)src[i];
}

// ---------------------------------------------------------------------------
// Pack weights W[z][K][Ncols] (fp32 row-major) into bf16 WMMA B-fragments.
// Fragment element j for lane L of tile (z,tileN,kt):
//   col  = tileN*16 + (L&15), koff = (L<16)?0:8
//   j<8 : k = kt*32 + koff + j ; j>=8 : k = kt*32 + 16 + koff + (j-8)
// Stored contiguously: Bp[(((z*NT + tileN)*KT + kt)*32 + L)*16 + j]
// ---------------------------------------------------------------------------
__global__ void pack_B_kernel(const float* __restrict__ W, bf16* __restrict__ Bp,
                              int K, int Ncols, int zCount) {
    const int NTt = Ncols >> 4;
    const int KT  = K >> 5;
    long total = (long)zCount * NTt * KT * 32;
    long i = (long)blockIdx.x * blockDim.x + threadIdx.x;
    if (i >= total) return;
    int lane = (int)(i & 31);
    long t   = i >> 5;
    int kt   = (int)(t % KT);   t /= KT;
    int tileN= (int)(t % NTt);  t /= NTt;
    int z    = (int)t;

    const int sub  = lane & 15;
    const int koff = (lane < 16) ? 0 : 8;
    const int col  = tileN * 16 + sub;
    const int kb   = kt * 32;
    const float* Wz = W + (long)z * K * Ncols;
    bf16* out = Bp + ((((long)z * NTt + tileN) * KT + kt) * 32 + lane) * 16;
#pragma unroll
    for (int j = 0; j < 8; ++j) {
        out[j]     = (bf16)Wz[(long)(kb + koff + j)      * Ncols + col];
        out[j + 8] = (bf16)Wz[(long)(kb + 16 + koff + j) * Ncols + col];
    }
}

// ---------------------------------------------------------------------------
// relbias[r*HC+hc] = sum_d rel_emb[r,d] * W_e[r, De+d, hc]
// ---------------------------------------------------------------------------
__global__ void relbias_kernel(const float* __restrict__ rel_emb,
                               const float* __restrict__ W_e,
                               float* __restrict__ relbias,
                               int R, int Dr, int De, int HC) {
    int i = blockIdx.x * blockDim.x + threadIdx.x;
    if (i >= R * HC) return;
    int r = i / HC, hc = i - r * HC;
    float s = 0.f;
    for (int d = 0; d < Dr; ++d)
        s += rel_emb[r * Dr + d] * W_e[((long)r * (De + Dr) + De + d) * (long)HC + hc];
    relbias[i] = s;
}

// ---------------------------------------------------------------------------
// init: segmax = monotone-int(-inf), segsum = 0, aggC = 0
// ---------------------------------------------------------------------------
__global__ void init_seg_kernel(int* __restrict__ segmax, float* __restrict__ segsum,
                                float* __restrict__ aggC, long nSeg, long nAgg) {
    long i = (long)blockIdx.x * blockDim.x + threadIdx.x;
    long stride = (long)gridDim.x * blockDim.x;
    for (long j = i; j < nSeg; j += stride) { segmax[j] = (int)0x80000000; segsum[j] = 0.f; }
    for (long j = i; j < nAgg; j += stride) aggC[j] = 0.f;
}

// ---------------------------------------------------------------------------
// bf16 WMMA GEMM with pre-packed B fragments and async-LDS-staged A panel.
//   D[z] = A (MxK) * B[z] (KxN) + bias[z]
// Block: 128 threads = 4 waves. Each wave owns 2 adjacent 16-col tiles
// (2 accumulators, 2 v_wmma per K-step). A panel [16 x K] staged once per
// block into LDS via global_load_async_to_lds_b128 (ASYNCcnt path).
// ---------------------------------------------------------------------------
#define GEMM_WAVES 4
#define GEMM_NT    2
#define GEMM_KMAX  256

template <bool OUT_BF16, int ACT>   // ACT: 0 = none, 1 = SiLU
__global__ void gemm_bf16_wmma(const bf16* __restrict__ A,
                               const bf16* __restrict__ Bp0,
                               const float* __restrict__ bias0,
                               void* __restrict__ D0,
                               int M, int K, int Ncols,
                               int lda, int ldd,
                               long sBp, long sD, long sBias) {
    __shared__ bf16 As[16 * GEMM_KMAX];

    const int lane    = threadIdx.x & 31;
    const int wave    = threadIdx.x >> 5;
    const int z       = blockIdx.z;
    const int rowBase = blockIdx.x * 16;

    // ---- stage A panel [16][K] into LDS (16-byte chunks, async) ----
    {
        const int chunksPerRow = K >> 3;           // 8 bf16 per 16B
        const int nChunk = 16 * chunksPerRow;
        for (int ch = threadIdx.x; ch < nChunk; ch += blockDim.x) {
            int row  = ch / chunksPerRow;
            int col8 = ch - row * chunksPerRow;
            int arow = rowBase + row; if (arow >= M) arow = M - 1;
            const bf16* gp = A + (long)arow * lda + col8 * 8;
            bf16* lp = As + row * K + col8 * 8;
#if USE_ASYNC
            unsigned ldsOff = (unsigned)(uintptr_t)lp;       // low 32 bits = LDS byte addr
            unsigned long long ga = (unsigned long long)(uintptr_t)gp;
            asm volatile("global_load_async_to_lds_b128 %0, %1, off"
                         :: "v"(ldsOff), "v"(ga) : "memory");
#else
            *(v8bf*)lp = *(const v8bf*)gp;
#endif
        }
#if USE_ASYNC
        asm volatile("s_wait_asynccnt 0x0" ::: "memory");
#endif
        __syncthreads();
    }

    const int NTt    = Ncols >> 4;
    const int KT     = K >> 5;
    const int tileN0 = (blockIdx.y * GEMM_WAVES + wave) * GEMM_NT;
    if (tileN0 >= NTt) return;
    const int tileN1 = (tileN0 + 1 < NTt) ? (tileN0 + 1) : tileN0;
    const bool has1  = (tileN0 + 1 < NTt);

    const int sub  = lane & 15;
    const int koff = (lane < 16) ? 0 : 8;
    const bf16* Bz  = Bp0 + (long)z * sBp;
    const bf16* bp0 = Bz + (((long)tileN0 * KT) * 32 + lane) * 16;
    const bf16* bp1 = Bz + (((long)tileN1 * KT) * 32 + lane) * 16;

    v8f c0 = {}, c1 = {};
#pragma unroll 2
    for (int kt = 0; kt < KT; ++kt) {
        const bf16* asp = As + sub * K + kt * 32 + koff;
        v8bf a0 = *(const v8bf*)asp;           // ds_load_b128
        v8bf a1 = *(const v8bf*)(asp + 16);
        v16bf a, b0, b1;
        b0 = *(const v16bf*)(bp0 + (long)kt * 32 * 16);  // contiguous 32B fragment
        b1 = *(const v16bf*)(bp1 + (long)kt * 32 * 16);
#pragma unroll
        for (int i = 0; i < 8; ++i) { a[i] = a0[i]; a[i + 8] = a1[i]; }
        c0 = __builtin_amdgcn_wmma_f32_16x16x32_bf16(false, a, false, b0, (short)0, c0, false, false);
        c1 = __builtin_amdgcn_wmma_f32_16x16x32_bf16(false, a, false, b1, (short)0, c1, false, false);
    }

    const float* bias = bias0 ? (bias0 + (long)z * sBias) : nullptr;
    const int colA = tileN0 * 16 + sub;
    const int colB = tileN1 * 16 + sub;
    const float bvA = bias ? bias[colA] : 0.0f;
    const float bvB = bias ? bias[colB] : 0.0f;
    const int rOff = (lane < 16) ? 0 : 8;
    bf16*  Db = (bf16*)D0  + (long)z * sD;
    float* Df = (float*)D0 + (long)z * sD;
#pragma unroll
    for (int i = 0; i < 8; ++i) {
        int row = rowBase + rOff + i;
        if (row < M) {
            float vA = c0[i] + bvA;
            if (ACT == 1) vA = vA / (1.0f + __expf(-vA));   // SiLU
            if (OUT_BF16) Db[(long)row * ldd + colA] = (bf16)vA;
            else          Df[(long)row * ldd + colA] = vA;
            if (has1) {
                float vB = c1[i] + bvB;
                if (ACT == 1) vB = vB / (1.0f + __expf(-vB));
                if (OUT_BF16) Db[(long)row * ldd + colB] = (bf16)vB;
                else          Df[(long)row * ldd + colB] = vB;
            }
        }
    }
}

// ---------------------------------------------------------------------------
// Per-edge GATv2 score: one wave32 per edge.
// ---------------------------------------------------------------------------
__global__ void edge_score_kernel(const bf16* __restrict__ XL, const bf16* __restrict__ XR,
                                  const float* __restrict__ edge_attr,
                                  const int* __restrict__ edge_index,
                                  const int* __restrict__ edge_type,
                                  const float* __restrict__ W_e,
                                  const float* __restrict__ relbias,
                                  const float* __restrict__ att,
                                  float* __restrict__ score, int* __restrict__ segmax,
                                  int E, int R, int H, int C, int HC, int De, int Dr) {
    int e = blockIdx.x * (blockDim.x >> 5) + (threadIdx.x >> 5);
    if (e >= E) return;
    const int lane = threadIdx.x & 31;
    const int r = edge_type[e];
    const int s = edge_index[e];
    const int d = edge_index[E + e];

    const long xlb = ((long)s * R + r) * HC;
    const long xrb = ((long)d * R + r) * HC;
    const long web = (long)r * (De + Dr) * HC;

    float ea[16];
    for (int j = 0; j < De; ++j) ea[j] = edge_attr[(long)e * De + j];

    float acc[8];
#pragma unroll
    for (int hh = 0; hh < 8; ++hh) acc[hh] = 0.f;

    for (int hc = lane; hc < HC; hc += 32) {
        float ef = relbias[r * HC + hc];
        for (int j = 0; j < De; ++j)
            ef += ea[j] * W_e[web + (long)j * HC + hc];
        float v = (float)XL[xlb + hc] + (float)XR[xrb + hc] + ef;
        v = (v > 0.f) ? v : NEG_SLOPE * v;
        int hh = hc / C;
        acc[hh] += v * att[((long)r * H + hh) * C + (hc - hh * C)];
    }

    const int seg = d * R + r;
    for (int hh = 0; hh < H; ++hh) {
        float ssum = acc[hh];
        for (int off = 16; off > 0; off >>= 1)
            ssum += __shfl_xor(ssum, off, 32);               // wave32 reduction
        if (lane == 0) {
            score[(long)e * H + hh] = ssum;
            int bits = __float_as_int(ssum);
            bits = (bits >= 0) ? bits : (bits ^ 0x7fffffff); // monotone f32->int map
            atomicMax(&segmax[(long)seg * H + hh], bits);
        }
    }
}

// ---------------------------------------------------------------------------
// alpha = exp(score - segmax); segsum += alpha
// ---------------------------------------------------------------------------
__global__ void alpha_kernel(const float* __restrict__ score, const int* __restrict__ segmax,
                             const int* __restrict__ edge_index, const int* __restrict__ edge_type,
                             float* __restrict__ alpha, float* __restrict__ segsum,
                             int E, int R, int H) {
    int i = blockIdx.x * blockDim.x + threadIdx.x;
    if (i >= E * H) return;
    int e = i / H, hh = i - e * H;
    int r = edge_type[e];
    int d = edge_index[E + e];
    int seg = d * R + r;
    int mb = segmax[(long)seg * H + hh];
    float m = __int_as_float((mb >= 0) ? mb : (mb ^ 0x7fffffff));
    float a = __expf(score[i] - m);
    alpha[i] = a;
    atomicAdd(&segsum[(long)seg * H + hh], a);
}

// ---------------------------------------------------------------------------
// aggC[dst,r,c] += mean_h( alpha_norm[e,h] * xl_e[e,h*C+c] )
// ---------------------------------------------------------------------------
__global__ void aggregate_kernel(const bf16* __restrict__ XL,
                                 const float* __restrict__ alpha,
                                 const float* __restrict__ segsum,
                                 const int* __restrict__ edge_index,
                                 const int* __restrict__ edge_type,
                                 float* __restrict__ aggC,
                                 int E, int R, int H, int C, int HC) {
    int e = blockIdx.x * (blockDim.x >> 5) + (threadIdx.x >> 5);
    if (e >= E) return;
    const int lane = threadIdx.x & 31;
    const int r = edge_type[e];
    const int s = edge_index[e];
    const int d = edge_index[E + e];
    const int seg = d * R + r;

    float an[8];
    for (int hh = 0; hh < H; ++hh)
        an[hh] = alpha[(long)e * H + hh] / segsum[(long)seg * H + hh];
    const float invH = 1.0f / (float)H;

    const long xlb = ((long)s * R + r) * HC;
    const long ab  = ((long)d * R + r) * C;
    for (int c = lane; c < C; c += 32) {
        float v = 0.f;
        for (int hh = 0; hh < H; ++hh)
            v += an[hh] * (float)XL[xlb + hh * C + c];
        atomicAdd(&aggC[ab + c], v * invH);
    }
}

// ---------------------------------------------------------------------------
// h1 = LN(h + sum_r softmax(gate)[r]*(aggC[:,r,:]+bias[r]))  (wave per row)
// ---------------------------------------------------------------------------
__global__ void fuse_ln1_kernel(const float* __restrict__ h, const float* __restrict__ aggC,
                                const float* __restrict__ bias, const float* __restrict__ gate,
                                const float* __restrict__ g1, const float* __restrict__ bt1,
                                float* __restrict__ h1, bf16* __restrict__ h1b,
                                int Nn, int F, int R, int C) {
    int n = blockIdx.x * (blockDim.x >> 5) + (threadIdx.x >> 5);
    if (n >= Nn) return;
    const int lane = threadIdx.x & 31;

    float gw[8];
    float gm = -1e30f;
    for (int r = 0; r < R; ++r) gm = fmaxf(gm, gate[r]);
    float gs = 0.f;
    for (int r = 0; r < R; ++r) { gw[r] = __expf(gate[r] - gm); gs += gw[r]; }
    for (int r = 0; r < R; ++r) gw[r] /= gs;

    float xv[8];
    float sum = 0.f, sq = 0.f;
    int cnt = 0;
    for (int c = lane; c < F; c += 32, ++cnt) {
        float hm = 0.f;
        for (int r = 0; r < R; ++r)
            hm += gw[r] * (aggC[((long)n * R + r) * C + c] + bias[r * C + c]);
        float x = h[(long)n * F + c] + hm;
        xv[cnt] = x;
        sum += x; sq += x * x;
    }
    for (int off = 16; off > 0; off >>= 1) {
        sum += __shfl_xor(sum, off, 32);
        sq  += __shfl_xor(sq,  off, 32);
    }
    float mu   = sum / F;
    float var  = sq / F - mu * mu;
    float rstd = rsqrtf(var + EPS);
    cnt = 0;
    for (int c = lane; c < F; c += 32, ++cnt) {
        float y = (xv[cnt] - mu) * rstd * g1[c] + bt1[c];
        h1 [(long)n * F + c] = y;
        h1b[(long)n * F + c] = (bf16)y;
    }
}

// ---------------------------------------------------------------------------
// out = LN(h1 + ffn2)  (wave per row)
// ---------------------------------------------------------------------------
__global__ void ln2_kernel(const float* __restrict__ h1, const float* __restrict__ ffn2,
                           const float* __restrict__ g2, const float* __restrict__ bt2,
                           float* __restrict__ out, int Nn, int F) {
    int n = blockIdx.x * (blockDim.x >> 5) + (threadIdx.x >> 5);
    if (n >= Nn) return;
    const int lane = threadIdx.x & 31;
    float xv[8];
    float sum = 0.f, sq = 0.f;
    int cnt = 0;
    for (int c = lane; c < F; c += 32, ++cnt) {
        float x = h1[(long)n * F + c] + ffn2[(long)n * F + c];
        xv[cnt] = x;
        sum += x; sq += x * x;
    }
    for (int off = 16; off > 0; off >>= 1) {
        sum += __shfl_xor(sum, off, 32);
        sq  += __shfl_xor(sq,  off, 32);
    }
    float mu   = sum / F;
    float var  = sq / F - mu * mu;
    float rstd = rsqrtf(var + EPS);
    cnt = 0;
    for (int c = lane; c < F; c += 32, ++cnt)
        out[(long)n * F + c] = (xv[cnt] - mu) * rstd * g2[c] + bt2[c];
}

// ---------------------------------------------------------------------------
// Host-side launcher
// ---------------------------------------------------------------------------
extern "C" void kernel_launch(void* const* d_in, const int* in_sizes, int n_in,
                              void* d_out, int out_size, void* d_ws, size_t ws_size,
                              hipStream_t stream) {
    const float* h          = (const float*)d_in[0];
    const int*   edge_index = (const int*)  d_in[1];
    const float* edge_attr  = (const float*)d_in[2];
    const int*   edge_type  = (const int*)  d_in[3];
    const float* rel_emb    = (const float*)d_in[4];
    const float* W_l        = (const float*)d_in[5];
    const float* b_l        = (const float*)d_in[6];
    const float* W_r        = (const float*)d_in[7];
    const float* b_r        = (const float*)d_in[8];
    const float* W_e        = (const float*)d_in[9];
    const float* att        = (const float*)d_in[10];
    const float* bias       = (const float*)d_in[11];
    const float* gate       = (const float*)d_in[12];
    const float* g1         = (const float*)d_in[13];
    const float* bt1        = (const float*)d_in[14];
    const float* g2         = (const float*)d_in[15];
    const float* bt2        = (const float*)d_in[16];
    const float* Wf1        = (const float*)d_in[17];
    const float* bf1        = (const float*)d_in[18];
    const float* Wf2        = (const float*)d_in[19];
    const float* bf2        = (const float*)d_in[20];

    const int F   = in_sizes[13];          // g1 : [F]
    const int R   = in_sizes[12];          // gate : [R]
    const int E   = in_sizes[3];           // edge_type : [E]
    const int N   = in_sizes[0] / F;       // h : [N,F]
    const int De  = in_sizes[2] / E;       // edge_attr : [E,De]
    const int Dr  = in_sizes[4] / R;       // rel_emb : [R,Dr]
    const int HC  = in_sizes[6] / R;       // b_l : [R,HC]
    const int C   = in_sizes[11] / R;      // bias : [R,C]
    const int H   = HC / C;
    const int FFN = in_sizes[18];          // bf1 : [FFN]

    // --- carve workspace ---
    char*  base = (char*)d_ws;
    size_t off  = 0;
    auto carve = [&](size_t bytes) -> void* {
        void* p = base + off;
        off += (bytes + 255) & ~(size_t)255;
        return p;
    };
    bf16*  hb      = (bf16*) carve((size_t)N * F * 2);
    bf16*  WlP     = (bf16*) carve((size_t)R * F * HC * 2);   // packed fragments
    bf16*  WrP     = (bf16*) carve((size_t)R * F * HC * 2);
    bf16*  Wf1P    = (bf16*) carve((size_t)F * FFN * 2);
    bf16*  Wf2P    = (bf16*) carve((size_t)FFN * F * 2);
    float* relbias = (float*)carve((size_t)R * HC * 4);
    bf16*  XL      = (bf16*) carve((size_t)N * R * HC * 2);
    bf16*  XR      = (bf16*) carve((size_t)N * R * HC * 2);
    float* score   = (float*)carve((size_t)E * H * 4);
    float* alpha   = (float*)carve((size_t)E * H * 4);
    int*   segmax  = (int*)  carve((size_t)N * R * H * 4);
    float* segsum  = (float*)carve((size_t)N * R * H * 4);
    float* aggC    = (float*)carve((size_t)N * R * C * 4);
    float* h1      = (float*)carve((size_t)N * F * 4);
    bf16*  h1b     = (bf16*) carve((size_t)N * F * 2);
    bf16*  ffn1b   = (bf16*) carve((size_t)N * FFN * 2);
    float* ffn2    = (float*)carve((size_t)N * F * 4);

    const int T = 256;
    auto gsz = [&](long n) -> int {
        long b = (n + T - 1) / T;
        return (int)((b > 65535) ? 65535 : b);
    };

    // 1) convert h; pack all weight matrices into WMMA fragment layout
    cvt_f32_bf16_kernel<<<gsz((long)N * F), T, 0, stream>>>(h, hb, (long)N * F);
    pack_B_kernel<<<gsz((long)R * (HC / 16) * (F / 32) * 32), T, 0, stream>>>(W_l, WlP, F, HC, R);
    pack_B_kernel<<<gsz((long)R * (HC / 16) * (F / 32) * 32), T, 0, stream>>>(W_r, WrP, F, HC, R);
    pack_B_kernel<<<gsz((long)(FFN / 16) * (F / 32) * 32),    T, 0, stream>>>(Wf1, Wf1P, F, FFN, 1);
    pack_B_kernel<<<gsz((long)(F / 16) * (FFN / 32) * 32),    T, 0, stream>>>(Wf2, Wf2P, FFN, F, 1);

    // 2) relation embedding bias
    relbias_kernel<<<gsz((long)R * HC), T, 0, stream>>>(rel_emb, W_e, relbias, R, Dr, De, HC);

    // 3) init segment buffers
    init_seg_kernel<<<gsz((long)N * R * C), T, 0, stream>>>(
        segmax, segsum, aggC, (long)N * R * H, (long)N * R * C);

    // 4) per-relation projections: XL = h@W_l[r] + b_l[r], XR = h@W_r[r] + b_r[r]
    {
        const int colsPerBlk = GEMM_WAVES * GEMM_NT * 16;     // 128
        dim3 blk(GEMM_WAVES * 32);
        dim3 grd((N + 15) / 16, (HC + colsPerBlk - 1) / colsPerBlk, R);
        gemm_bf16_wmma<true, 0><<<grd, blk, 0, stream>>>(
            hb, WlP, b_l, XL, N, F, HC, F, R * HC,
            (long)F * HC, (long)HC, (long)HC);
        gemm_bf16_wmma<true, 0><<<grd, blk, 0, stream>>>(
            hb, WrP, b_r, XR, N, F, HC, F, R * HC,
            (long)F * HC, (long)HC, (long)HC);
    }

    // 5) attention scores + segment max
    edge_score_kernel<<<(E + 7) / 8, 256, 0, stream>>>(
        XL, XR, edge_attr, edge_index, edge_type, W_e, relbias, att,
        score, segmax, E, R, H, C, HC, De, Dr);

    // 6) alpha = exp(score - max); segment sum
    alpha_kernel<<<gsz((long)E * H), T, 0, stream>>>(
        score, segmax, edge_index, edge_type, alpha, segsum, E, R, H);

    // 7) aggregate messages (head mean)
    aggregate_kernel<<<(E + 7) / 8, 256, 0, stream>>>(
        XL, alpha, segsum, edge_index, edge_type, aggC, E, R, H, C, HC);

    // 8) gated fusion + LayerNorm1
    fuse_ln1_kernel<<<(N + 7) / 8, 256, 0, stream>>>(
        h, aggC, bias, gate, g1, bt1, h1, h1b, N, F, R, C);

    // 9) FFN: silu(h1@Wf1 + bf1) @ Wf2 + bf2
    {
        const int colsPerBlk = GEMM_WAVES * GEMM_NT * 16;
        dim3 blk(GEMM_WAVES * 32);
        dim3 grd1((N + 15) / 16, (FFN + colsPerBlk - 1) / colsPerBlk, 1);
        gemm_bf16_wmma<true, 1><<<grd1, blk, 0, stream>>>(
            h1b, Wf1P, bf1, ffn1b, N, F, FFN, F, FFN, 0, 0, 0);
        dim3 grd2((N + 15) / 16, (F + colsPerBlk - 1) / colsPerBlk, 1);
        gemm_bf16_wmma<false, 0><<<grd2, blk, 0, stream>>>(
            ffn1b, Wf2P, bf2, ffn2, N, FFN, F, FFN, F, 0, 0, 0);
    }

    // 10) residual + LayerNorm2 -> output
    ln2_kernel<<<(N + 7) / 8, 256, 0, stream>>>(h1, ffn2, g2, bt2, (float*)d_out, N, F);
}
